// Net_27101243638138
// MI455X (gfx1250) — matrix operand order repaired and checked by reading, not compiled, not run on hardware
//
#include <hip/hip_runtime.h>
#include <math.h>

// ---------------- types ----------------
typedef __attribute__((ext_vector_type(16))) _Float16 v16h;
typedef __attribute__((ext_vector_type(8)))  float    v8f;

#define KSZ 15           // spline kernel size per dim
#define NEG_BIG -3.402823466e38f

// ---------------- small utility kernels ----------------
__global__ void fill_kernel(float* __restrict__ p, float v, int n) {
    int i = blockIdx.x * blockDim.x + threadIdx.x;
    if (i < n) p[i] = v;
}

__device__ __forceinline__ void atomicMaxFloat(float* addr, float val) {
    // init value must be negative (we use -FLT_MAX)
    if (val >= 0.f) atomicMax((int*)addr, __float_as_int(val));
    else            atomicMin((unsigned int*)addr, (unsigned int)__float_as_int(val));
}

// segment_max pooling: out[cluster[i], c] = max(h[i, c])
__global__ void pool_max_kernel(const float* __restrict__ h, const int* __restrict__ cluster,
                                float* __restrict__ out, int nrows) {
    int idx = blockIdx.x * blockDim.x + threadIdx.x;
    if (idx >= nrows * 64) return;
    int i = idx >> 6, c = idx & 63;
    atomicMaxFloat(&out[(size_t)cluster[i] * 64 + c], h[idx]);
}

// hin[i, 0..63] = h[i, 0..63]; hin[i, 64] = 1
__global__ void build_hin_kernel(const float* __restrict__ h, float* __restrict__ hin, int n) {
    int idx = blockIdx.x * blockDim.x + threadIdx.x;
    if (idx >= n * 65) return;
    int i = idx / 65, c = idx % 65;
    hin[idx] = (c < 64) ? h[(size_t)i * 64 + c] : 1.0f;
}

// ---------------- edge gather-weight-scatter (spline message passing) ----------------
// 4 edges per 256-thread block; lane group of 64 threads covers the 64 output channels.
template <int CIN>
__global__ void edge_conv_kernel(const float* __restrict__ X,      // [Nin, CIN]
                                 const int*   __restrict__ src,
                                 const int*   __restrict__ dst,
                                 const float* __restrict__ pseudo, // [E, 3]
                                 const float* __restrict__ W,      // [3375, CIN, 64]
                                 float* __restrict__ acc,          // [Nout, 64] (zeroed)
                                 float* __restrict__ deg,          // [Nout]     (zeroed)
                                 int E) {
    __shared__ float xs[4][CIN];
    __shared__ float bb[4][8];
    __shared__ int   kk[4][8];
    __shared__ int   dd[4];
    __shared__ int   sv[4];

    const int tid = threadIdx.x;
    const int e0  = blockIdx.x * 4;

    // stage x[src] for the 4 edges
    for (int idx = tid; idx < 4 * CIN; idx += 256) {
        int el = idx / CIN, c = idx % CIN;
        int e = e0 + el;
        float v = 0.f;
        if (e < E) v = X[(size_t)src[e] * CIN + c];
        xs[el][c] = v;
    }
    // one thread per edge: basis taps + kernel indices
    if (tid < 4) {
        int e = e0 + tid;
        sv[tid] = (e < E);
        if (e < E) {
            dd[tid] = dst[e];
            float fr[3]; int lo[3];
#pragma unroll
            for (int d = 0; d < 3; ++d) {
                float u = pseudo[(size_t)e * 3 + d] * (float)(KSZ - 1);
                float l = floorf(u);
                l = fminf(fmaxf(l, 0.f), (float)(KSZ - 2));
                fr[d] = u - l;
                lo[d] = (int)l;
            }
            const int pw[3] = {1, KSZ, KSZ * KSZ};
#pragma unroll
            for (int s = 0; s < 8; ++s) {
                float b = 1.f; int k = 0;
#pragma unroll
                for (int d = 0; d < 3; ++d) {
                    int bit = (s >> d) & 1;
                    b *= bit ? fr[d] : (1.f - fr[d]);
                    k += (lo[d] + bit) * pw[d];
                }
                bb[tid][s] = b; kk[tid][s] = k;
            }
        }
    }
    __syncthreads();

    const int el = tid >> 6;
    const int o  = tid & 63;
    if (!sv[el]) return;

    float m = 0.f;
#pragma unroll
    for (int s = 0; s < 8; ++s) {
        // prefetch next tap's weight column into cache while computing this one
        if (s + 1 < 8)
            __builtin_prefetch(W + (size_t)kk[el][s + 1] * (CIN * 64) + o, 0, 0);
        const float bsv = bb[el][s];
        const float* __restrict__ Wk = W + (size_t)kk[el][s] * (CIN * 64) + o;
        float t = 0.f;
#pragma unroll 4
        for (int c = 0; c < CIN; ++c) t = fmaf(xs[el][c], Wk[(size_t)c * 64], t);
        m = fmaf(bsv, t, m);
    }
    atomicAdd(&acc[(size_t)dd[el] * 64 + o], m);
    if (o == 0) atomicAdd(&deg[dd[el]], 1.f);
}

// ---------------- WMMA dense root GEMM: R[N,64] = X[N,CIN] @ Wr[CIN,64] ----------------
// 4 waves per 128-thread block; each wave owns one 16x16 output tile (n0 = wave*16).
// A and B are staged in LDS *pre-swizzled into WMMA fragment order* so each lane
// loads its 16 halves as one contiguous 32-byte chunk (2x ds_load_b128, no shuffles).
// N is always a multiple of 16 here, so stores are unguarded (EXEC stays all-ones).
template <int CIN>
__global__ void root_gemm_kernel(const float* __restrict__ X,   // [N, CIN] row-major
                                 const float* __restrict__ Wr,  // [CIN, 64]
                                 float* __restrict__ R) {       // [N, 64]
    constexpr int TILES = (CIN + 31) / 32;
    constexpr int KPAD  = TILES * 32;

    __shared__ __align__(32) _Float16 AsF[TILES * 32 * 16];      // [kt][lane][h]
    __shared__ __align__(32) _Float16 BsF[TILES * 4 * 32 * 16];  // [kt][wave][lane][h]

    const int tid = threadIdx.x;
    const int m0  = blockIdx.x * 16;

    // stage A (16 rows x KPAD) directly in fragment order
    for (int idx = tid; idx < 16 * KPAD; idx += 128) {
        int r = idx / KPAD, k = idx % KPAD;
        float v = (k < CIN) ? X[(size_t)(m0 + r) * CIN + k] : 0.f;
        int kt  = k >> 5, kk = k & 31;
        int grp = (kk >> 3) & 1;                    // A: K 8..15 / 24..31 -> lane group 1
        int hh  = (kk & 7) + ((kk >> 4) & 1) * 8;   //    K 16..31 -> upper 8 halves
        int lane = r + grp * 16;
        AsF[(kt * 32 + lane) * 16 + hh] = (_Float16)v;
    }
    // stage B (KPAD x 64) in fragment order, one copy per wave's 16-column tile
    for (int idx = tid; idx < KPAD * 64; idx += 128) {
        int k = idx / 64, n = idx % 64;
        float v = (k < CIN) ? Wr[(size_t)k * 64 + n] : 0.f;
        int kt  = k >> 5, kk = k & 31;
        int wv  = n >> 4, nn = n & 15;
        int grp = (kk >> 3) & 1;                    // B: K 8..15 / 24..31 -> lane group 1
        int hh  = (kk & 7) + ((kk >> 4) & 1) * 8;   //    K 16..31 -> upper 8 halves
        int lane = nn + grp * 16;
        BsF[(((kt * 4) + wv) * 32 + lane) * 16 + hh] = (_Float16)v;
    }
    __syncthreads();

    const int wave   = tid >> 5;
    const int lane   = tid & 31;
    const int lane15 = lane & 15;
    const int grp    = lane >> 4;
    const int n0     = wave * 16;

    v8f c = {};
#pragma unroll
    for (int kt = 0; kt < TILES; ++kt) {
        v16h a = *(const v16h*)&AsF[(kt * 32 + lane) * 16];
        v16h b = *(const v16h*)&BsF[(((kt * 4) + wave) * 32 + lane) * 16];
        c = __builtin_amdgcn_wmma_f32_16x16x32_f16(
                /*neg_a=*/false, a, /*neg_b=*/false, b,
                /*c_mod=*/(short)0, c, /*reuse_a=*/false, /*reuse_b=*/false);
    }
    // C/D layout: VGPR r -> row m0 + r + grp*8, col n0 + lane%16
#pragma unroll
    for (int r = 0; r < 8; ++r)
        R[(size_t)(m0 + r + grp * 8) * 64 + n0 + lane15] = c[r];
}

// ---------------- conv epilogue: out = f(acc/deg + R + bias [+ resid]) ----------------
__global__ void epilogue_kernel(const float* __restrict__ acc, const float* __restrict__ deg,
                                const float* __restrict__ R, const float* __restrict__ bias,
                                const float* __restrict__ resid, float* __restrict__ out,
                                int n, int do_elu) {
    int idx = blockIdx.x * blockDim.x + threadIdx.x;
    if (idx >= n * 64) return;
    int i = idx >> 6, c = idx & 63;
    float d = deg[i]; d = (d < 1.f) ? 1.f : d;
    float v = acc[idx] / d + R[idx] + bias[c];
    if (resid) v += resid[idx];
    if (do_elu) v = (v > 0.f) ? v : expm1f(v);
    out[idx] = v;
}

// ---------------- batchnorm ----------------
__global__ void bn_stats_kernel(const float* __restrict__ h, int n, float* __restrict__ stats) {
    int c  = threadIdx.x & 63;
    int rp = blockDim.x >> 6;
    int r0 = blockIdx.x * rp + (threadIdx.x >> 6);
    float s = 0.f, q = 0.f;
    for (int i = r0; i < n; i += gridDim.x * rp) {
        float v = h[(size_t)i * 64 + c];
        s += v; q += v * v;
    }
    atomicAdd(&stats[c], s);
    atomicAdd(&stats[64 + c], q);
}

__global__ void bn_apply_kernel(float* __restrict__ h, int n, const float* __restrict__ stats,
                                const float* __restrict__ g, const float* __restrict__ b) {
    int idx = blockIdx.x * blockDim.x + threadIdx.x;
    if (idx >= n * 64) return;
    int c = idx & 63;
    float inv_n = 1.f / (float)n;
    float mean = stats[c] * inv_n;
    float var  = stats[64 + c] * inv_n - mean * mean;
    h[idx] = g[c] * (h[idx] - mean) * rsqrtf(var + 1e-5f) + b[c];
}

// ---------------- readout: segment mean over batch ----------------
__global__ void seg_sum_kernel(const float* __restrict__ h, const int* __restrict__ batch,
                               float* __restrict__ gsum, float* __restrict__ cnt, int n) {
    int idx = blockIdx.x * blockDim.x + threadIdx.x;
    if (idx >= n * 64) return;
    int i = idx >> 6, c = idx & 63;
    atomicAdd(&gsum[(size_t)batch[i] * 64 + c], h[idx]);
    if (c == 0) atomicAdd(&cnt[batch[i]], 1.f);
}

__global__ void fc_head_kernel(const float* __restrict__ gsum, const float* __restrict__ cnt,
                               const float* __restrict__ fcW, const float* __restrict__ fcb,
                               float* __restrict__ out) {
    __shared__ float L[16 * 40];
    int tid = threadIdx.x;
    if (tid < 640) {
        int b = tid / 40, j = tid % 40;
        float inv = 1.f / fmaxf(cnt[b], 1.f);
        float s = fcb[j];
#pragma unroll 8
        for (int c = 0; c < 64; ++c) s = fmaf(gsum[b * 64 + c] * inv, fcW[c * 40 + j], s);
        L[tid] = s;
    }
    __syncthreads();
    if (tid < 640) {
        int b = tid / 40;
        float mx = NEG_BIG;
        for (int j = 0; j < 40; ++j) mx = fmaxf(mx, L[b * 40 + j]);
        float se = 0.f;
        for (int j = 0; j < 40; ++j) se += expf(L[b * 40 + j] - mx);
        out[tid] = L[tid] - mx - logf(se);
    }
}

// ---------------- host orchestration ----------------
static inline int ceil_div(int a, int b) { return (a + b - 1) / b; }

extern "C" void kernel_launch(void* const* d_in, const int* in_sizes, int n_in,
                              void* d_out, int out_size, void* d_ws, size_t ws_size,
                              hipStream_t stream) {
    (void)in_sizes; (void)n_in; (void)out_size; (void)ws_size;

    static const int NS[7] = {16384, 8192, 4096, 2048, 1024, 512, 256};
    const int KT = KSZ * KSZ * KSZ;   // 3375

    const float* x = (const float*)d_in[0];
    const int*   srcs[7]; const int* dsts[7]; const float* pseudos[7];
    for (int l = 0; l < 7; ++l) {
        srcs[l]    = (const int*)  d_in[1 + 3 * l];
        dsts[l]    = (const int*)  d_in[2 + 3 * l];
        pseudos[l] = (const float*)d_in[3 + 3 * l];
    }
    const int* clusters[6];
    for (int l = 0; l < 6; ++l) clusters[l] = (const int*)d_in[22 + l];
    const int*   batch = (const int*)  d_in[28];
    const float* W1    = (const float*)d_in[29];
    const float* root1 = (const float*)d_in[30];
    const float* b1    = (const float*)d_in[31];
    const float* WA    = (const float*)d_in[32];
    const float* rootA = (const float*)d_in[33];
    const float* biasA = (const float*)d_in[34];
    const float* WB    = (const float*)d_in[35];
    const float* rootB = (const float*)d_in[36];
    const float* biasB = (const float*)d_in[37];
    const float* bng   = (const float*)d_in[38];
    const float* bnb   = (const float*)d_in[39];
    const float* fcW   = (const float*)d_in[40];
    const float* fcb   = (const float*)d_in[41];
    float* out = (float*)d_out;

    // ---- workspace bump allocator ----
    char* wp = (char*)d_ws;
    auto alloc = [&](size_t nfloats) -> float* {
        float* p = (float*)wp;
        wp += ((nfloats * sizeof(float) + 255) / 256) * 256;
        return p;
    };
    float* hA    = alloc((size_t)16384 * 64);
    float* hB    = alloc((size_t)16384 * 64);
    float* hin   = alloc((size_t)8192 * 65);
    float* accb  = alloc((size_t)16384 * 64);
    float* degb  = alloc(16384);
    float* Rb    = alloc((size_t)16384 * 64);
    float* tb    = alloc((size_t)8192 * 64);
    float* stats = alloc(128);
    float* gsum  = alloc(16 * 64);
    float* cntb  = alloc(16);

    auto fill = [&](float* p, float v, int n) {
        fill_kernel<<<ceil_div(n, 256), 256, 0, stream>>>(p, v, n);
    };
    auto run_bn = [&](float* h, int n, const float* g, const float* b) {
        fill(stats, 0.f, 128);
        bn_stats_kernel<<<64, 256, 0, stream>>>(h, n, stats);
        bn_apply_kernel<<<ceil_div(n * 64, 256), 256, 0, stream>>>(h, n, stats, g, b);
    };

    float* h   = hA;
    float* alt = hB;

    // ---- level 0 conv (1 -> 64) ----
    {
        const int N = NS[0], E = 12 * NS[0];
        fill(accb, 0.f, N * 64);
        fill(degb, 0.f, N);
        edge_conv_kernel<1><<<ceil_div(E, 4), 256, 0, stream>>>(
            x, srcs[0], dsts[0], pseudos[0], W1, accb, degb, E);
        root_gemm_kernel<1><<<N / 16, 128, 0, stream>>>(x, root1, Rb);
        epilogue_kernel<<<ceil_div(N * 64, 256), 256, 0, stream>>>(
            accb, degb, Rb, b1, nullptr, h, N, /*elu=*/1);
        run_bn(h, N, bng + 0 * 64, bnb + 0 * 64);
    }

    // ---- 6 residual levels ----
    for (int l = 0; l < 6; ++l) {
        const int Nin = NS[l], N = NS[l + 1], E = 12 * NS[l + 1];

        // voxel max pool: h (Nin rows) -> alt (N rows)
        fill(alt, NEG_BIG, N * 64);
        pool_max_kernel<<<ceil_div(Nin * 64, 256), 256, 0, stream>>>(h, clusters[l], alt, Nin);
        { float* tmp = h; h = alt; alt = tmp; }   // h now has N rows

        // hin = [h, 1]
        build_hin_kernel<<<ceil_div(N * 65, 256), 256, 0, stream>>>(h, hin, N);

        // conv A: 65 -> 64, ELU
        fill(accb, 0.f, N * 64);
        fill(degb, 0.f, N);
        edge_conv_kernel<65><<<ceil_div(E, 4), 256, 0, stream>>>(
            hin, srcs[l + 1], dsts[l + 1], pseudos[l + 1],
            WA + (size_t)l * KT * 65 * 64, accb, degb, E);
        root_gemm_kernel<65><<<N / 16, 128, 0, stream>>>(
            hin, rootA + (size_t)l * 65 * 64, Rb);
        epilogue_kernel<<<ceil_div(N * 64, 256), 256, 0, stream>>>(
            accb, degb, Rb, biasA + (size_t)l * 64, nullptr, tb, N, /*elu=*/1);

        // conv B: 64 -> 64, + residual h, ELU
        fill(accb, 0.f, N * 64);
        fill(degb, 0.f, N);
        edge_conv_kernel<64><<<ceil_div(E, 4), 256, 0, stream>>>(
            tb, srcs[l + 1], dsts[l + 1], pseudos[l + 1],
            WB + (size_t)l * KT * 64 * 64, accb, degb, E);
        root_gemm_kernel<64><<<N / 16, 128, 0, stream>>>(
            tb, rootB + (size_t)l * 64 * 64, Rb);
        epilogue_kernel<<<ceil_div(N * 64, 256), 256, 0, stream>>>(
            accb, degb, Rb, biasB + (size_t)l * 64, /*resid=*/h, h, N, /*elu=*/1);

        run_bn(h, N, bng + (size_t)(l + 1) * 64, bnb + (size_t)(l + 1) * 64);
    }

    // ---- readout: batch mean -> FC -> log_softmax ----
    fill(gsum, 0.f, 16 * 64);
    fill(cntb, 0.f, 16);
    seg_sum_kernel<<<ceil_div(NS[6] * 64, 256), 256, 0, stream>>>(h, batch, gsum, cntb, NS[6]);
    fc_head_kernel<<<1, 640, 0, stream>>>(gsum, cntb, fcW, fcb, out);
}
